// ImprovedPolicyHead_60318520705092
// MI455X (gfx1250) — matrix operand order repaired and checked by reading, not compiled, not run on hardware
//
#include <hip/hip_runtime.h>
#include <hip/hip_bf16.h>
#include <math.h>

typedef __attribute__((ext_vector_type(16))) _Float16 v16h;
typedef __attribute__((ext_vector_type(8)))  _Float16 v8h;
typedef __attribute__((ext_vector_type(4)))  _Float16 v4h;
typedef __attribute__((ext_vector_type(8)))  float    v8f;
typedef __attribute__((ext_vector_type(4)))  float    v4f;

union V16 { v16h v; v8h h[2]; };

#define LN_EPS 1e-5f

__device__ __forceinline__ float gelu_exact(float x) {
    return 0.5f * x * (1.0f + erff(x * 0.7071067811865476f));
}

// ---------------------------------------------------------------------------
// Weight prep: cast f32 -> f16 and transpose so B-matrix fragments become
// two contiguous 8-half per-lane chunks ([N][K] row-major).
// ---------------------------------------------------------------------------
__global__ void prep_weights_kernel(const float* __restrict__ W1,
                                    const float* __restrict__ W2,
                                    _Float16* __restrict__ w1t,
                                    _Float16* __restrict__ w2t) {
    const int H = 128, H2 = 64;
    int t = blockIdx.x * blockDim.x + threadIdx.x;
    if (t < H * H) {                       // w1t[n][k] = W1[k][n]
        int n = t >> 7, k = t & 127;
        w1t[t] = (_Float16)W1[(size_t)k * H + n];
    }
    if (t < H2 * H) {                      // w2t[n][k] = W2[k][n]
        int n = t >> 7, k = t & 127;
        w2t[t] = (_Float16)W2[(size_t)k * H2 + n];
    }
}

// ---------------------------------------------------------------------------
// Fused LayerNorm + MLP (WMMA f16->f32) + scatter of scores.
// 8 waves / block, each wave owns 16 edges (rows).
// Layer 1 runs in two halves of 4 N-tiles to cap live accumulator VGPRs at 32
// (avoids scratch spills); each half drains into a second LDS buffer lB.
// ---------------------------------------------------------------------------
__global__ __launch_bounds__(256)
void policy_mlp_kernel(const float* __restrict__ ef,
                       const int*   __restrict__ srcI,
                       const int*   __restrict__ dstI,
                       const float* __restrict__ ln_g,
                       const float* __restrict__ ln_b,
                       const _Float16* __restrict__ w1t,
                       const float* __restrict__ b1,
                       const _Float16* __restrict__ w2t,
                       const float* __restrict__ b2,
                       const float* __restrict__ W3,
                       const float* __restrict__ b3,
                       float* __restrict__ scores,
                       long long numEdges, int N, int E) {
    constexpr int H = 128;
    __shared__ _Float16 lA[8][16][H] __attribute__((aligned(16)));  // LN'd input
    __shared__ _Float16 lB[8][16][H] __attribute__((aligned(16)));  // layer-1 out

    const int wave = threadIdx.x >> 5;
    const int lane = threadIdx.x & 31;
    const int nlo  = lane & 15;
    const int hi   = lane >> 4;
    const long long rowBase = ((long long)blockIdx.x * 8 + wave) * 16;

    // ---- LayerNorm: 16 rows, 4 f32 per lane, wave-wide reduction ----------
    const float4 gv = *(const float4*)(ln_g + lane * 4);
    const float4 bv = *(const float4*)(ln_b + lane * 4);
    for (int r = 0; r < 16; ++r) {
        const long long e = rowBase + r;          // uniform per wave
        v4f x = (v4f)(0.0f);
        if (e < numEdges)
            x = __builtin_nontemporal_load(
                (const v4f*)(ef + (size_t)e * H + lane * 4));   // streaming read
        float s  = x.x + x.y + x.z + x.w;
        float s2 = x.x * x.x + x.y * x.y + x.z * x.z + x.w * x.w;
        for (int m = 1; m < 32; m <<= 1) {
            s  += __shfl_xor(s,  m, 32);
            s2 += __shfl_xor(s2, m, 32);
        }
        const float mu  = s * (1.0f / H);
        const float var = s2 * (1.0f / H) - mu * mu;
        const float rs  = rsqrtf(var + LN_EPS);
        v4h hv;
        hv[0] = (_Float16)((x.x - mu) * rs * gv.x + bv.x);
        hv[1] = (_Float16)((x.y - mu) * rs * gv.y + bv.y);
        hv[2] = (_Float16)((x.z - mu) * rs * gv.z + bv.z);
        hv[3] = (_Float16)((x.w - mu) * rs * gv.w + bv.w);
        *(v4h*)&lA[wave][r][lane * 4] = hv;
    }

    const _Float16* arowA = &lA[wave][nlo][0];
    const _Float16* arowB = &lB[wave][nlo][0];

    // ---- Layer 1: [16x128] x [128x128], two halves of 4 N-tiles -----------
    for (int half = 0; half < 2; ++half) {
        v8f acc[4] = {};
        for (int ko = 0; ko < 4; ++ko) {
            V16 A;
            const int c0 = ko * 32 + hi * 8;      // A layout: two 8-half chunks
            A.h[0] = *(const v8h*)(arowA + c0);
            A.h[1] = *(const v8h*)(arowA + c0 + 16);
            for (int ntl = 0; ntl < 4; ++ntl) {
                const int nt = half * 4 + ntl;
                V16 B;
                const _Float16* bp =
                    w1t + (size_t)(nlo + nt * 16) * H + ko * 32 + hi * 8;
                B.h[0] = *(const v8h*)(bp);
                B.h[1] = *(const v8h*)(bp + 16);
                acc[ntl] = __builtin_amdgcn_wmma_f32_16x16x32_f16(
                    false, A.v, false, B.v, (short)0, acc[ntl], false, false);
            }
        }
        // bias + exact GELU, drain into lB (frees the 32 acc VGPRs)
        for (int ntl = 0; ntl < 4; ++ntl) {
            const int n = nlo + (half * 4 + ntl) * 16;
            const float bias = b1[n];
            for (int v = 0; v < 8; ++v) {
                const float y = gelu_exact(acc[ntl][v] + bias);
                lB[wave][v + hi * 8][n] = (_Float16)y;
            }
        }
    }

    // ---- Layer 2: [16x128] x [128x64]  (4 K-steps x 4 N-tiles) ------------
    v8f acc2[4] = {};
    for (int ko = 0; ko < 4; ++ko) {
        V16 A;
        const int c0 = ko * 32 + hi * 8;
        A.h[0] = *(const v8h*)(arowB + c0);
        A.h[1] = *(const v8h*)(arowB + c0 + 16);
        for (int nt = 0; nt < 4; ++nt) {
            V16 B;
            const _Float16* bp =
                w2t + (size_t)(nlo + nt * 16) * H + ko * 32 + hi * 8;
            B.h[0] = *(const v8h*)(bp);
            B.h[1] = *(const v8h*)(bp + 16);
            acc2[nt] = __builtin_amdgcn_wmma_f32_16x16x32_f16(
                false, A.v, false, B.v, (short)0, acc2[nt], false, false);
        }
    }

    // ---- bias + GELU + 64->1 projection (VALU dot + butterfly reduce) -----
    float part[8] = {};
    for (int nt = 0; nt < 4; ++nt) {
        const int n = nlo + nt * 16;
        const float bias = b2[n];
        const float w3v  = W3[n];
        for (int v = 0; v < 8; ++v) {
            const float y = gelu_exact(acc2[nt][v] + bias);
            part[v] += y * w3v;
        }
    }
    for (int v = 0; v < 8; ++v)
        for (int m = 1; m < 16; m <<= 1)
            part[v] += __shfl_xor(part[v], m, 32);   // reduce within 16-lane halves

    if (nlo == 0) {                                   // lanes 0 (rows 0-7) and 16 (rows 8-15)
        const float b3v = b3[0];
        for (int v = 0; v < 8; ++v) {
            const long long e = rowBase + v + hi * 8;
            if (e < numEdges) {
                const int sN = srcI[e], dN = dstI[e];
                const int g  = sN / N;
                const int ls = sN - g * N, ld = dN - g * N;
                const int i  = ls < ld ? ls : ld;
                const int j  = ls < ld ? ld : ls;
                if (i != j) {
                    const int idx = i * N - (i * (i + 1)) / 2 + (j - i - 1);
                    scores[(size_t)g * E + idx] = part[v] + b3v;
                }
            }
        }
    }
}

// ---------------------------------------------------------------------------
// Row softmax over E slots per graph.
// ---------------------------------------------------------------------------
__global__ __launch_bounds__(256)
void softmax_kernel(const float* __restrict__ scores,
                    float* __restrict__ out, int E) {
    const int g   = blockIdx.x;
    const int tid = threadIdx.x;
    const float* sp = scores + (size_t)g * E;
    float* op = out + (size_t)g * E;
    __shared__ float red[256];

    float mx = -3.402823466e+38f;
    for (int e = tid; e < E; e += 256) mx = fmaxf(mx, sp[e]);
    red[tid] = mx; __syncthreads();
    for (int s = 128; s > 0; s >>= 1) {
        if (tid < s) red[tid] = fmaxf(red[tid], red[tid + s]);
        __syncthreads();
    }
    mx = red[0]; __syncthreads();

    float sum = 0.f;
    for (int e = tid; e < E; e += 256) sum += expf(sp[e] - mx);
    red[tid] = sum; __syncthreads();
    for (int s = 128; s > 0; s >>= 1) {
        if (tid < s) red[tid] += red[tid + s];
        __syncthreads();
    }
    const float inv = 1.0f / red[0];
    for (int e = tid; e < E; e += 256)
        __builtin_nontemporal_store(expf(sp[e] - mx) * inv, op + e);
}

// ---------------------------------------------------------------------------
extern "C" void kernel_launch(void* const* d_in, const int* in_sizes, int n_in,
                              void* d_out, int out_size, void* d_ws, size_t ws_size,
                              hipStream_t stream) {
    const float* ef   = (const float*)d_in[0];
    const int*   eidx = (const int*)  d_in[1];
    const float* ln_g = (const float*)d_in[4];
    const float* ln_b = (const float*)d_in[5];
    const float* W1   = (const float*)d_in[6];
    const float* b1   = (const float*)d_in[7];
    const float* W2   = (const float*)d_in[8];
    const float* b2   = (const float*)d_in[9];
    const float* W3   = (const float*)d_in[10];
    const float* b3   = (const float*)d_in[11];

    const int H = 128;
    const long long numEdges = (long long)in_sizes[0] / H;
    const int G = in_sizes[3];
    const int N = in_sizes[2] / G;
    const int E = (int)(numEdges / G);

    _Float16* w1t = (_Float16*)d_ws;
    _Float16* w2t = (_Float16*)((char*)d_ws + 128 * 128 * 2);
    float*    scr = (float*)   ((char*)d_ws + 128 * 128 * 2 + 64 * 128 * 2);

    const int* srcI = eidx;
    const int* dstI = eidx + numEdges;

    prep_weights_kernel<<<64, 256, 0, stream>>>(W1, W2, w1t, w2t);

    const int blocks = (int)((numEdges + 127) / 128);
    policy_mlp_kernel<<<blocks, 256, 0, stream>>>(
        ef, srcI, dstI, ln_g, ln_b, w1t, b1, w2t, b2, W3, b3,
        scr, numEdges, N, E);

    softmax_kernel<<<G, 256, 0, stream>>>(scr, (float*)d_out, E);
}
